// HybridAttentionQuantum_65481071395575
// MI455X (gfx1250) — matrix operand order repaired and checked by reading, not compiled, not run on hardware
//
#include <hip/hip_runtime.h>
#include <hip/hip_bf16.h>

// ---------------------------------------------------------------------------
// HybridAttentionQuantum on MI455X (gfx1250, wave32, WMMA f16 16x16x32 + TDM)
//
//   stage 1: Q/K/V projections (x @ W^T). W tile TDM-staged to double-
//            buffered LDS (overlap DMA/compute); each wave computes a 32x64
//            output tile (2 m-subtiles reuse B frags in registers).
//            V stored transposed (b,h,dk,s).
//   stage 2: flash attention: K/V tiles TDM-staged into double-buffered LDS
//            (shared by 8 waves of one (b,h)), QK^T + P@V via WMMA, online
//            softmax via shfl over 16-lane halves, P transposed through LDS.
//            Epilogue fuses /l, 0.5*cos(x+phi) quantum branch, 0.5 mix.
//   stage 3: mixed(f16) @ Wo^T -> f32 out (one GEMM for both branches),
//            same double-buffered TDM structure as stage 1.
// ---------------------------------------------------------------------------

#define B_  4
#define S_  2048
#define E_  1024
#define H_  16
#define DK_ 64

typedef __attribute__((ext_vector_type(16))) _Float16     v16h;
typedef __attribute__((ext_vector_type(8)))  float        v8f;
typedef __attribute__((ext_vector_type(4)))  unsigned int u32x4;
typedef __attribute__((ext_vector_type(4)))  int          i32x4;
typedef __attribute__((ext_vector_type(8)))  int          i32x8;

#if defined(__AMDGCN__) && __has_builtin(__builtin_amdgcn_tensor_load_to_lds)
#define HAVE_TDM 1
#else
#define HAVE_TDM 0
#endif

#if HAVE_TDM
#if __has_include(<hip/amd_detail/amd_gfx1250_TDM.h>)
// therock/clang-23 toolchain: 6-arg form (extra i32x8 group)
#define TDM_CALL(g0, g1, g2, g3) \
  __builtin_amdgcn_tensor_load_to_lds(g0, g1, g2, g3, \
      ((i32x8){0, 0, 0, 0, 0, 0, 0, 0}), 0)
#else
// ROCm 7.2 / clang-22: 5-arg form (confirmed lowering this round)
#define TDM_CALL(g0, g1, g2, g3) \
  __builtin_amdgcn_tensor_load_to_lds(g0, g1, g2, g3, 0)
#endif
#endif

// ---- TDM 2D tile load: tensor row-major, tile_x contiguous elements, ------
// ---- tile_y rows separated by stride_x elements; compacted into LDS. ------
template <typename T>
__device__ inline void tdm_load_tile(T* ldsPtr, const T* gptr, unsigned tile_x,
                                     unsigned tile_y, unsigned stride_x) {
#if HAVE_TDM
  constexpr unsigned dsz =
      (sizeof(T) == 1) ? 0u : (sizeof(T) == 2) ? 1u : (sizeof(T) == 4) ? 2u : 3u;
  const unsigned long long ga = (unsigned long long)(uintptr_t)gptr;
  const unsigned lds_off = (unsigned)(uintptr_t)ldsPtr;  // LDS aperture low32
  u32x4 g0;
  g0[0] = 1u;                                  // count=1 user descriptor
  g0[1] = lds_off;                             // lds_addr (bytes)
  g0[2] = (unsigned)(ga & 0xFFFFFFFFu);        // global_addr[31:0]
  g0[3] = (unsigned)((ga >> 32) & 0x1FFFFFFu)  // global_addr[56:32]
          | (2u << 30);                        // type=2 (image)
  const unsigned tdim = 0x40000000u;           // huge tensor dims: no OOB clip
  i32x8 g1;
  g1[0] = (int)(dsz << 16);                        // wg_mask=0, data_size
  g1[1] = (int)(tdim << 16);                       // tensor_dim0[15:0]
  g1[2] = (int)((tdim >> 16) | (tdim << 16));      // dim0 hi / dim1 lo
  g1[3] = (int)((tdim >> 16) | (tile_x << 16));    // dim1 hi / tile_dim0
  g1[4] = (int)(tile_y & 0xFFFFu);                 // tile_dim1, tile_dim2=0
  g1[5] = (int)stride_x;                           // tensor_dim0_stride[31:0]
  g1[6] = 0;                                       // stride0 hi, stride1 lo
  g1[7] = 0;
  const i32x4 z4 = {0, 0, 0, 0};
  TDM_CALL(g0, g1, z4, z4);
#else
  // fallback: wave-cooperative copy (caller gates to one wave)
  const int lane = threadIdx.x & 31;
  for (unsigned y = 0; y < tile_y; ++y)
    for (unsigned xi = lane; xi < tile_x; xi += 32)
      ldsPtr[y * tile_x + xi] = gptr[(size_t)y * stride_x + xi];
#endif
}

__device__ inline void tdm_wait() {
#if HAVE_TDM
  __builtin_amdgcn_s_wait_tensorcnt(0);
#endif
}

// --- fragment loaders ------------------------------------------------------
// 16-bit A fragment (16x32): lane r=lane&15 holds row M=r; half h=lane>>4:
// elems 0..15 -> K = 16h + i contiguous
__device__ inline v16h a_frag_from_f32(const float* row, int half) {
  const float* p0 = row + half * 16;
  const float* p1 = row + 8 + half * 16;
  v16h a;
#pragma unroll
  for (int i = 0; i < 8; ++i) a[i] = (_Float16)p0[i];
#pragma unroll
  for (int i = 0; i < 8; ++i) a[8 + i] = (_Float16)p1[i];
  return a;
}
__device__ inline v16h a_frag_from_f16(const _Float16* row, int half) {
  const _Float16* p0 = row + half * 16;
  const _Float16* p1 = row + 8 + half * 16;
  v16h a;
#pragma unroll
  for (int i = 0; i < 8; ++i) a[i] = p0[i];
#pragma unroll
  for (int i = 0; i < 8; ++i) a[8 + i] = p1[i];
  return a;
}
// 16-bit B fragment (32x16): lane n=lane&15 is column; K = 16h + i contiguous.
__device__ inline v16h b_frag_from_f32(const float* p) {
  v16h b;
#pragma unroll
  for (int i = 0; i < 16; ++i) b[i] = (_Float16)p[i];
  return b;
}
__device__ inline v16h b_frag_from_f16(const _Float16* p) {
  v16h b;
#pragma unroll
  for (int i = 0; i < 16; ++i) b[i] = p[i];
  return b;
}

// --- stage 1: QKV projection ----------------------------------------------
// grid: (1024, 3) x 128thr. 4 waves/block; wave tile = 32 rows x 64 cols
// (2 m-subtiles). Block spans 4 consecutive heads -> W tile 256 rows x 32 k,
// TDM double-buffered in LDS.
__global__ __launch_bounds__(128) void qkv_proj_kernel(
    const float* __restrict__ x, const float* __restrict__ Wq,
    const float* __restrict__ Wk, const float* __restrict__ Wv,
    _Float16* __restrict__ qo, _Float16* __restrict__ ko,
    _Float16* __restrict__ vto) {
  __shared__ float ldsW[2][256 * 32];  // 2 x 32 KB staged weight tiles

  const int mode = blockIdx.y;  // 0=Q 1=K 2=V
  const float* __restrict__ W = (mode == 0) ? Wq : (mode == 1) ? Wk : Wv;

  const int lane = threadIdx.x & 31;
  const int wave = threadIdx.x >> 5;
  const int tile = blockIdx.x * 4 + wave;  // 0..4095
  const int mPair = tile >> 4;             // 256 pairs of 16-row tiles
  const int head  = tile & 15;
  const int m0 = mPair * 32;
  const int r = lane & 15;
  const int half = lane >> 4;
  const int blockN0 = ((blockIdx.x * 4) & 15) * 64;  // first W row of block

  v8f acc[2][4] = {};
  const float* __restrict__ xrow0 = x + (size_t)(m0 + r) * E_;
  const float* __restrict__ xrow1 = x + (size_t)(m0 + 16 + r) * E_;

  if (wave == 0) tdm_load_tile(ldsW[0], W + (size_t)blockN0 * E_, 32u, 256u, E_);

  for (int kk = 0, buf = 0; kk < E_; kk += 32, buf ^= 1) {
    if (wave == 0) tdm_wait();
    __syncthreads();
    if (wave == 0 && kk + 32 < E_)
      tdm_load_tile(ldsW[buf ^ 1], W + (size_t)blockN0 * E_ + kk + 32, 32u,
                    256u, E_);
    v16h a0 = a_frag_from_f32(xrow0 + kk, half);
    v16h a1 = a_frag_from_f32(xrow1 + kk, half);
    __builtin_prefetch(xrow0 + kk + 32, 0, 3);
    __builtin_prefetch(xrow1 + kk + 32, 0, 3);
#pragma unroll
    for (int sub = 0; sub < 4; ++sub) {
      const float* wrow =
          &ldsW[buf][(wave * 64 + sub * 16 + r) * 32 + half * 16];
      v16h bfrag = b_frag_from_f32(wrow);  // shared across both m-subtiles
      acc[0][sub] = __builtin_amdgcn_wmma_f32_16x16x32_f16(
          false, a0, false, bfrag, (short)0, acc[0][sub], false, false);
      acc[1][sub] = __builtin_amdgcn_wmma_f32_16x16x32_f16(
          false, a1, false, bfrag, (short)0, acc[1][sub], false, false);
    }
    __syncthreads();
  }

  // C layout: element j -> row M = j + 8*half, col N = r (within subtile).
  const int b_idx = m0 / S_;
  const int s0 = m0 % S_;
  const size_t bh = (size_t)b_idx * H_ + head;
#pragma unroll
  for (int mt = 0; mt < 2; ++mt) {
#pragma unroll
    for (int sub = 0; sub < 4; ++sub) {
      const int d = sub * 16 + r;
#pragma unroll
      for (int j = 0; j < 8; ++j) {
        const int srow = s0 + mt * 16 + j + half * 8;
        const _Float16 v = (_Float16)acc[mt][sub][j];
        if (mode == 0)      qo[(bh * S_ + srow) * DK_ + d] = v;
        else if (mode == 1) ko[(bh * S_ + srow) * DK_ + d] = v;
        else                vto[(bh * DK_ + d) * S_ + srow] = v;  // transposed
      }
    }
  }
}

// --- stage 2: flash attention + quantum mix -------------------------------
// grid: (B*H, S/16/8) x 256thr; one wave = 16 query rows of one (b,h).
// K/V tiles for the whole block TDM-staged into double-buffered LDS.
__global__ __launch_bounds__(256) void flash_attn_kernel(
    const float* __restrict__ x, const _Float16* __restrict__ q,
    const _Float16* __restrict__ k, const _Float16* __restrict__ vt,
    const float* __restrict__ phi, _Float16* __restrict__ mixed) {
  __shared__ _Float16 ldsK[2][32 * DK_];  // 2 x 4 KB: 32 keys x 64 dk
  __shared__ _Float16 ldsV[2][DK_ * 32];  // 2 x 4 KB: 64 dk x 32 keys
  __shared__ _Float16 ldsP[8][16 * 32];   // per-wave P transpose staging

  const int lane = threadIdx.x & 31;
  const int wave = threadIdx.x >> 5;
  const int bh = blockIdx.x;  // 0..63
  const int b_idx = bh / H_;
  const int head  = bh % H_;
  const int qtile = blockIdx.y * 8 + wave;
  const int s0 = qtile * 16;
  const int r = lane & 15;
  const int half = lane >> 4;

  const _Float16* __restrict__ qbase = q  + (size_t)bh * S_ * DK_;
  const _Float16* __restrict__ kbase = k  + (size_t)bh * S_ * DK_;
  const _Float16* __restrict__ vbase = vt + (size_t)bh * DK_ * S_;

  // Q fragments, two dk-chunks of 32, prescaled by 1/sqrt(dk)=0.125
  v16h qa[2];
  {
    const _Float16* qrow = qbase + (size_t)(s0 + r) * DK_;
#pragma unroll
    for (int c = 0; c < 2; ++c) {
#pragma unroll
      for (int i = 0; i < 8; ++i)
        qa[c][i] = (_Float16)((float)qrow[c * 32 + half * 16 + i] * 0.125f);
#pragma unroll
      for (int i = 0; i < 8; ++i)
        qa[c][8 + i] =
            (_Float16)((float)qrow[c * 32 + 8 + half * 16 + i] * 0.125f);
    }
  }

  float row_m[8], row_l[8];
#pragma unroll
  for (int j = 0; j < 8; ++j) {
    row_m[j] = -__builtin_inff();
    row_l[j] = 0.0f;
  }
  v8f acc[4] = {};
  _Float16* pbuf = ldsP[wave];

  // prologue: stage first K/V tile
  if (wave == 0) {
    tdm_load_tile(ldsK[0], kbase, 32u * DK_, 1u, 32u * DK_);
    tdm_load_tile(ldsV[0], vbase, 32u, (unsigned)DK_, S_);
  }

  for (int kb = 0, buf = 0; kb < S_; kb += 32, buf ^= 1) {
    if (wave == 0) tdm_wait();  // current buffer resident
    __syncthreads();
    if (wave == 0 && kb + 32 < S_) {  // prefetch next tile during compute
      tdm_load_tile(ldsK[buf ^ 1], kbase + (size_t)(kb + 32) * DK_,
                    32u * DK_, 1u, 32u * DK_);
      tdm_load_tile(ldsV[buf ^ 1], vbase + (kb + 32), 32u, (unsigned)DK_, S_);
    }

    // ---- scores for 32 keys: two 16-key subtiles, each = 2 chained WMMAs
    v8f sc[2];
#pragma unroll
    for (int ks = 0; ks < 2; ++ks) {
      const _Float16* krow = &ldsK[buf][(ks * 16 + r) * DK_ + half * 16];
      v16h bk0 = b_frag_from_f16(krow);       // dk 0..31 slice
      v16h bk1 = b_frag_from_f16(krow + 32);  // dk 32..63 slice
      v8f s = {};
      s = __builtin_amdgcn_wmma_f32_16x16x32_f16(false, qa[0], false, bk0,
                                                 (short)0, s, false, false);
      s = __builtin_amdgcn_wmma_f32_16x16x32_f16(false, qa[1], false, bk1,
                                                 (short)0, s, false, false);
      sc[ks] = s;
    }

    // ---- online softmax: rows live in elements, cols in 16-lane halves
    float scale[8];
#pragma unroll
    for (int j = 0; j < 8; ++j) {
      float bm = fmaxf(sc[0][j], sc[1][j]);
#pragma unroll
      for (int off = 1; off < 16; off <<= 1)
        bm = fmaxf(bm, __shfl_xor(bm, off, 16));
      const float mn = fmaxf(row_m[j], bm);
      scale[j] = __expf(row_m[j] - mn);
      row_m[j] = mn;
      const float p0 = __expf(sc[0][j] - mn);
      const float p1 = __expf(sc[1][j] - mn);
      sc[0][j] = p0;
      sc[1][j] = p1;
      float ps = p0 + p1;
#pragma unroll
      for (int off = 1; off < 16; off <<= 1) ps += __shfl_xor(ps, off, 16);
      row_l[j] = row_l[j] * scale[j] + ps;
    }

    // ---- transpose P (C layout) -> A layout via per-wave LDS tile
#pragma unroll
    for (int ks = 0; ks < 2; ++ks)
#pragma unroll
      for (int j = 0; j < 8; ++j)
        pbuf[(j + half * 8) * 32 + ks * 16 + r] = (_Float16)sc[ks][j];
    v16h pa = a_frag_from_f16(pbuf + r * 32, half);  // in-order per-wave LDS

    // ---- acc = acc*scale + P @ V  (V transposed: contiguous B frags)
#pragma unroll
    for (int sub = 0; sub < 4; ++sub) {
      v8f c = acc[sub];
#pragma unroll
      for (int j = 0; j < 8; ++j) c[j] *= scale[j];
      const _Float16* vrow = &ldsV[buf][(sub * 16 + r) * 32 + half * 16];
      v16h bv = b_frag_from_f16(vrow);
      acc[sub] = __builtin_amdgcn_wmma_f32_16x16x32_f16(
          false, pa, false, bv, (short)0, c, false, false);
    }
    __syncthreads();  // all waves done with ldsK/V[buf] before reuse
  }

  // ---- epilogue: mixed = 0.5*ctx/l + 0.5*cos(x + phi)
#pragma unroll
  for (int sub = 0; sub < 4; ++sub) {
    const int d = sub * 16 + r;
    const float ph = phi[d];
#pragma unroll
    for (int j = 0; j < 8; ++j) {
      const int srow = s0 + j + half * 8;
      const float ctx = acc[sub][j] / row_l[j];
      const float xv = x[((size_t)b_idx * S_ + srow) * E_ + head * 64 + d];
      const float mv = 0.5f * ctx + 0.5f * __cosf(xv + ph);
      mixed[((size_t)b_idx * S_ + srow) * E_ + head * 64 + d] = (_Float16)mv;
    }
  }
}

// --- stage 3: fused output projection: out = mixed @ Wo^T (f32 out) -------
// Same structure as stage 1: 32x64 per wave, double-buffered TDM W tile.
__global__ __launch_bounds__(128) void out_proj_kernel(
    const _Float16* __restrict__ mixed, const float* __restrict__ Wo,
    float* __restrict__ out) {
  __shared__ float ldsW[2][256 * 32];  // 2 x 32 KB staged Wo tiles

  const int lane = threadIdx.x & 31;
  const int wave = threadIdx.x >> 5;
  const int tile = blockIdx.x * 4 + wave;  // 0..4095
  const int mPair = tile >> 4;
  const int nTile = tile & 15;
  const int m0 = mPair * 32;
  const int n0 = nTile * 64;
  const int r = lane & 15;
  const int half = lane >> 4;
  const int blockN0 = ((blockIdx.x * 4) & 15) * 64;

  v8f acc[2][4] = {};
  const _Float16* __restrict__ arow0 = mixed + (size_t)(m0 + r) * E_;
  const _Float16* __restrict__ arow1 = mixed + (size_t)(m0 + 16 + r) * E_;

  if (wave == 0)
    tdm_load_tile(ldsW[0], Wo + (size_t)blockN0 * E_, 32u, 256u, E_);

  for (int kk = 0, buf = 0; kk < E_; kk += 32, buf ^= 1) {
    if (wave == 0) tdm_wait();
    __syncthreads();
    if (wave == 0 && kk + 32 < E_)
      tdm_load_tile(ldsW[buf ^ 1], Wo + (size_t)blockN0 * E_ + kk + 32, 32u,
                    256u, E_);
    v16h a0 = a_frag_from_f16(arow0 + kk, half);
    v16h a1 = a_frag_from_f16(arow1 + kk, half);
    __builtin_prefetch(arow0 + kk + 32, 0, 3);
    __builtin_prefetch(arow1 + kk + 32, 0, 3);
#pragma unroll
    for (int sub = 0; sub < 4; ++sub) {
      const float* wrow =
          &ldsW[buf][(wave * 64 + sub * 16 + r) * 32 + half * 16];
      v16h bfrag = b_frag_from_f32(wrow);
      acc[0][sub] = __builtin_amdgcn_wmma_f32_16x16x32_f16(
          false, a0, false, bfrag, (short)0, acc[0][sub], false, false);
      acc[1][sub] = __builtin_amdgcn_wmma_f32_16x16x32_f16(
          false, a1, false, bfrag, (short)0, acc[1][sub], false, false);
    }
    __syncthreads();
  }

#pragma unroll
  for (int mt = 0; mt < 2; ++mt)
#pragma unroll
    for (int sub = 0; sub < 4; ++sub)
#pragma unroll
      for (int j = 0; j < 8; ++j)
        out[(size_t)(m0 + mt * 16 + j + half * 8) * E_ + n0 + sub * 16 + r] =
            acc[mt][sub][j];
}

// ---------------------------------------------------------------------------
extern "C" void kernel_launch(void* const* d_in, const int* in_sizes, int n_in,
                              void* d_out, int out_size, void* d_ws,
                              size_t ws_size, hipStream_t stream) {
  const float* x   = (const float*)d_in[0];
  const float* Wq  = (const float*)d_in[1];
  const float* Wk  = (const float*)d_in[2];
  const float* Wv  = (const float*)d_in[3];
  const float* Wo  = (const float*)d_in[4];
  const float* phi = (const float*)d_in[5];
  float* out = (float*)d_out;

  const size_t per = (size_t)B_ * H_ * S_ * DK_;  // 8,388,608 halves = 16 MB
  _Float16* q     = (_Float16*)d_ws;
  _Float16* kbuf  = q + per;
  _Float16* vt    = kbuf + per;
  _Float16* mixed = vt + per;  // total 64 MB of f16 workspace

  dim3 g1(1024, 3);
  qkv_proj_kernel<<<g1, 128, 0, stream>>>(x, Wq, Wk, Wv, q, kbuf, vt);

  dim3 g2(B_ * H_, (S_ / 16) / 8);
  flash_attn_kernel<<<g2, 256, 0, stream>>>(x, q, kbuf, vt, phi, mixed);

  out_proj_kernel<<<1024, 128, 0, stream>>>(mixed, Wo, out);

  (void)in_sizes; (void)n_in; (void)out_size; (void)ws_size;
}
